// HybridGemmaDiT_42958262894760
// MI455X (gfx1250) — matrix operand order repaired and checked by reading, not compiled, not run on hardware
//
#include <hip/hip_runtime.h>
#include <hip/hip_bf16.h>
#include <math.h>

typedef __attribute__((ext_vector_type(16))) __bf16 v16bf;
typedef __attribute__((ext_vector_type(8)))  float  v8f;

#define DEV __device__ __forceinline__

constexpr float EPS = 1.1920929e-07f;

// ---- WMMA fragment helpers (wave32, 16x16x32 bf16) -------------------------
// A-matrix 16x32: lanes0-15 K=0..15, lanes16-31 K=16..31 (matches B).
DEV int frag_k(int lane, int e) {
  return ((lane >> 4) << 4) + e;
}

DEV v8f wmma_bf16(v16bf a, v16bf b, v8f c) {
  return __builtin_amdgcn_wmma_f32_16x16x32_bf16(false, a, false, b, (short)0, c,
                                                 false, false);
}

DEV v16bf afrag_bf16(const __bf16* row, int k0, int lane) {
  v16bf a;
#pragma unroll
  for (int e = 0; e < 16; ++e) a[e] = row[k0 + frag_k(lane, e)];
  return a;
}
// B-matrix 32x16: lanes0-15 K=0..15, lanes16-31 K=16..31. Source stored
// K-contiguous per column: baseT[n*ldk + k]  -> 32 contiguous bytes per frag.
DEV v16bf bfrag_bf16T(const __bf16* baseT, int ldk, int k0, int col, int lane) {
  const __bf16* row = baseT + (size_t)col * ldk + k0 + ((lane >> 4) << 4);
  v16bf b;
#pragma unroll
  for (int e = 0; e < 16; ++e) b[e] = row[e];
  return b;
}
// C/D: VGPR v -> M = m0 + v + 8*hi, N = col
DEV void cstore_f32(float* base, int ld, int m0_, int col, int lane, v8f c, float s) {
  int m0 = m0_ + ((lane >> 4) << 3);
#pragma unroll
  for (int v = 0; v < 8; ++v) base[(m0 + v) * ld + col] = c[v] * s;
}
DEV void cstore_bf16(__bf16* base, int ld, int m0_, int col, int lane, v8f c) {
  int m0 = m0_ + ((lane >> 4) << 3);
#pragma unroll
  for (int v = 0; v < 8; ++v) base[(m0 + v) * ld + col] = (__bf16)c[v];
}
// transposed store: base[col*ld + m] (8 contiguous bf16 per lane -> b128 store)
DEV void cstore_bf16T(__bf16* base, int ld, int m0_, int col, int lane, v8f c) {
  int m0 = m0_ + ((lane >> 4) << 3);
#pragma unroll
  for (int v = 0; v < 8; ++v) base[col * ld + m0 + v] = (__bf16)c[v];
}

// ---- block-level rmsnorm epilogue: LDS bf16 [64][264] -> global xn ---------
// 256 threads: 4 per token row; reads staged x, writes xn = bf16(x * rsqrt).
DEV void rms_epilogue(const __bf16 (*xs)[264], float* red, float* scale,
                      __bf16* xnb, int tid) {
  int r = tid >> 2, part = tid & 3;
  float s = 0.f;
#pragma unroll
  for (int i = 0; i < 64; ++i) {
    float v = (float)xs[r][part * 64 + i];
    s += v * v;
  }
  red[tid] = s;
  __syncthreads();
  if (part == 0) {
    float t = red[tid] + red[tid + 1] + red[tid + 2] + red[tid + 3];
    scale[r] = rsqrtf(t * (1.f / 256.f) + EPS);
  }
  __syncthreads();
  float sc = scale[r];
#pragma unroll
  for (int i = 0; i < 64; ++i)
    xnb[(size_t)r * 256 + part * 64 + i] = (__bf16)((float)xs[r][part * 64 + i] * sc);
}

// ---- weight convert: [K][N] f32 -> [N][K] bf16 (opt. per-K row scale) ------
__global__ __launch_bounds__(256) void wcvt_kernel(const float* in, __bf16* outT,
    int K, int N, const float* rowscale) {
  size_t total = (size_t)K * N;
  for (size_t idx = (size_t)blockIdx.x * 256 + threadIdx.x; idx < total;
       idx += (size_t)gridDim.x * 256) {
    int n = (int)(idx / K), k = (int)(idx % K);
    float v = in[(size_t)k * N + n];
    if (rowscale) v *= rowscale[k];
    outT[idx] = (__bf16)v;
  }
}

// ---- Householder product Q per layer -> bf16 Q and Q^T ---------------------
__global__ __launch_bounds__(64) void hh_kernel(const float* vs, __bf16* Qbf,
                                                __bf16* QTbf) {
  int l = blockIdx.x, j = threadIdx.x;
  __shared__ float Q[64][65];
  __shared__ float vsh[64];
  __shared__ float vn;
  for (int i = 0; i < 64; ++i) Q[i][j] = (i == j) ? 1.f : 0.f;
  __syncthreads();
  for (int r = 0; r < 32; ++r) {
    vsh[j] = vs[(size_t)l * 2048 + r * 64 + j];
    __syncthreads();
    float wj = 0.f;
    for (int i = 0; i < 64; ++i) wj += vsh[i] * Q[i][j];
    if (j == 0) {
      float s = 0.f;
      for (int i = 0; i < 64; ++i) s += vsh[i] * vsh[i];
      vn = s + 1e-8f;
    }
    __syncthreads();
    float c = 2.f / vn;
    for (int i = 0; i < 64; ++i) Q[i][j] -= c * vsh[i] * wj;
    __syncthreads();
  }
  for (int i = 0; i < 64; ++i) {
    Qbf[(size_t)l * 4096 + (size_t)i * 64 + j]  = (__bf16)Q[i][j];  // Q row-major
    QTbf[(size_t)l * 4096 + (size_t)j * 64 + i] = (__bf16)Q[i][j];  // Q transposed
  }
}

// ---- patchify + fourier + patch_in + fused rmsnorm -------------------------
__global__ __launch_bounds__(256) void embed_kernel(const float* z_t, const float* logsnr,
    const float* wpi, const float* bpi, float* xw, __bf16* xn) {
  int b = blockIdx.x, tid = threadIdx.x;
  __shared__ float  in20[64][20];
  __shared__ float  spin[8];
  __shared__ __bf16 xs[64][264];
  __shared__ float  red[256];
  __shared__ float  scale[64];
  for (int idx = tid; idx < 768; idx += 256) {
    int s = idx / 12, j = idx % 12;
    int gy = s >> 3, gx = s & 7, ch = j >> 2, py = (j >> 1) & 1, px = j & 1;
    in20[s][j] = z_t[((size_t)(b * 3 + ch) * 16 + gy * 2 + py) * 16 + gx * 2 + px];
  }
  if (tid < 8) {
    int q = tid & 3;
    float fr = 0.15707963267948966f * (float)(1 << q);  // 2*pi/40 * 2^q
    float a = logsnr[b] * fr;
    spin[tid] = (tid < 4) ? __sinf(a) : __cosf(a);
  }
  __syncthreads();
  for (int idx = tid; idx < 512; idx += 256) in20[idx >> 3][12 + (idx & 7)] = spin[idx & 7];
  __syncthreads();
  int c = tid;
  for (int s = 0; s < 64; ++s) {
    float sum = bpi[c];
#pragma unroll
    for (int k = 0; k < 20; ++k) sum += in20[s][k] * wpi[k * 256 + c];
    xw[((size_t)b * 64 + s) * 256 + c] = sum;
    xs[s][c] = (__bf16)sum;
  }
  __syncthreads();
  rms_epilogue(xs, red, scale, xn + (size_t)b * 64 * 256, tid);
}

// ---- attention (one block per batch*head) ----------------------------------
__global__ __launch_bounds__(256) void attn_kernel(const __bf16* xn,
    const __bf16* wqkvT_l, const __bf16* Qbf_l, const __bf16* QTbf_l,
    __bf16* ao, int gmask) {
  int b = blockIdx.x >> 2, h = blockIdx.x & 3;
  int tid = threadIdx.x, w = tid >> 5, lane = tid & 31, lr = lane & 15;
  __shared__ __bf16 qsc[64][72];   // rotation scratch (pre-rot / post-rope)
  __shared__ float  sf[64][68];    // q@Q^T result, later scores
  __shared__ __bf16 qb[64][72];    // rotated q, later attn weights
  __shared__ __bf16 kb[64][72];    // rotated k
  __shared__ __bf16 vb[64][72];    // v, stored transposed [dim][token]
  __shared__ float  red[64][4];
  const __bf16* xnb = xn + (size_t)b * 64 * 256;

  for (int which = 0; which < 3; ++which) {  // 0=q, 1=k, 2=v
    for (int t = w; t < 16; t += 8) {
      int mt = t >> 2, nt = t & 3, col = nt * 16 + lr;
      v8f acc = {};
      const __bf16* arow = xnb + (size_t)(mt * 16 + lr) * 256;
#pragma unroll
      for (int kb_ = 0; kb_ < 8; ++kb_) {
        int k0 = kb_ << 5;
        v16bf a  = afrag_bf16(arow, k0, lane);
        v16bf bm = bfrag_bf16T(wqkvT_l, 256, k0, which * 256 + h * 64 + col, lane);
        acc = wmma_bf16(a, bm, acc);
      }
      if (which == 2) cstore_bf16T(&vb[0][0], 72, mt * 16, col, lane, acc);
      else            cstore_bf16 (&qsc[0][0], 72, mt * 16, col, lane, acc);
    }
    __syncthreads();
    if (which < 2) {
      // sf = qsc @ Q^T   (B[k][n] = Q[n][k] = Qbf row-major, K-contiguous)
      for (int t = w; t < 16; t += 8) {
        int mt = t >> 2, nt = t & 3, col = nt * 16 + lr;
        v8f acc = {};
#pragma unroll
        for (int k2 = 0; k2 < 2; ++k2) {
          int k0 = k2 << 5;
          v16bf a  = afrag_bf16(&qsc[mt * 16 + lr][0], k0, lane);
          v16bf bm = bfrag_bf16T(Qbf_l, 64, k0, col, lane);
          acc = wmma_bf16(a, bm, acc);
        }
        cstore_f32(&sf[0][0], 68, mt * 16, col, lane, acc, 1.f);
      }
      __syncthreads();
      // RoPE (fp32 math) : qsc = sf*cos + rotate_half(sf)*sin  (bf16 store)
      for (int idx = tid; idx < 4096; idx += 256) {
        int s = idx >> 6, d = idx & 63;
        float val = sf[s][d];
        float prt = (d < 32) ? -sf[s][d + 32] : sf[s][d - 32];
        int j = d & 31;
        float coord = (j < 16) ? (float)(s >> 3) : (float)(s & 7);
        int fi = j & 15;
        float invf = __expf(-0.5756462732485115f * (float)fi);  // 10000^(-fi/16)
        float ang = coord * invf;
        qsc[s][d] = (__bf16)(val * __cosf(ang) + prt * __sinf(ang));
      }
      __syncthreads();
      // dst = qsc @ Q   (B[k][n] = Q[k][n] = QTbf[n*64+k], K-contiguous)
      __bf16* dst = (which == 0) ? &qb[0][0] : &kb[0][0];
      for (int t = w; t < 16; t += 8) {
        int mt = t >> 2, nt = t & 3, col = nt * 16 + lr;
        v8f acc = {};
#pragma unroll
        for (int k2 = 0; k2 < 2; ++k2) {
          int k0 = k2 << 5;
          v16bf a  = afrag_bf16(&qsc[mt * 16 + lr][0], k0, lane);
          v16bf bm = bfrag_bf16T(QTbf_l, 64, k0, col, lane);
          acc = wmma_bf16(a, bm, acc);
        }
        cstore_bf16(dst, 72, mt * 16, col, lane, acc);
      }
      __syncthreads();
    }
  }
  // scores = (qb @ kb^T)/8 -> sf   (B[k=d][n=tok] = kb[tok][d], K-contiguous)
  for (int t = w; t < 16; t += 8) {
    int mt = t >> 2, nt = t & 3, col = nt * 16 + lr;
    v8f acc = {};
#pragma unroll
    for (int k2 = 0; k2 < 2; ++k2) {
      int k0 = k2 << 5;
      v16bf a  = afrag_bf16(&qb[mt * 16 + lr][0], k0, lane);
      v16bf bm = bfrag_bf16T(&kb[0][0], 72, k0, col, lane);
      acc = wmma_bf16(a, bm, acc);
    }
    cstore_f32(&sf[0][0], 68, mt * 16, col, lane, acc, 0.125f);
  }
  __syncthreads();
  // masked softmax -> attn weights (bf16) into qb
  {
    int r = tid >> 2, part = tid & 3;
    float mx = -1e30f;
    for (int cc = part * 16; cc < part * 16 + 16; ++cc) {
      int dy = (r >> 3) - (cc >> 3), dx = (r & 7) - (cc & 7);
      if (gmask || (dy * dy + dx * dx <= 6)) mx = fmaxf(mx, sf[r][cc]);
    }
    red[r][part] = mx;
    __syncthreads();
    mx = fmaxf(fmaxf(red[r][0], red[r][1]), fmaxf(red[r][2], red[r][3]));
    __syncthreads();
    float sum = 0.f;
    for (int cc = part * 16; cc < part * 16 + 16; ++cc) {
      int dy = (r >> 3) - (cc >> 3), dx = (r & 7) - (cc & 7);
      bool ok = gmask || (dy * dy + dx * dx <= 6);
      float e = ok ? __expf(sf[r][cc] - mx) : 0.f;
      sf[r][cc] = e; sum += e;
    }
    red[r][part] = sum;
    __syncthreads();
    sum = red[r][0] + red[r][1] + red[r][2] + red[r][3];
    float inv = 1.f / sum;
    for (int cc = part * 16; cc < part * 16 + 16; ++cc)
      qb[r][cc] = (__bf16)(sf[r][cc] * inv);
  }
  __syncthreads();
  // out = attn @ v -> ao   (B[k=tok][n=dim] = vb[dim][tok], K-contiguous)
  for (int t = w; t < 16; t += 8) {
    int mt = t >> 2, nt = t & 3, col = nt * 16 + lr;
    v8f acc = {};
#pragma unroll
    for (int k2 = 0; k2 < 2; ++k2) {
      int k0 = k2 << 5;
      v16bf a  = afrag_bf16(&qb[mt * 16 + lr][0], k0, lane);
      v16bf bm = bfrag_bf16T(&vb[0][0], 72, k0, col, lane);
      acc = wmma_bf16(a, bm, acc);
    }
    int m0 = mt * 16 + ((lane >> 4) << 3);
#pragma unroll
    for (int v = 0; v < 8; ++v)
      ao[((size_t)b * 64 + m0 + v) * 256 + h * 64 + col] = (__bf16)acc[v];
  }
}

// ---- out-proj + gate + residual + fused rmsnorm ----------------------------
__global__ __launch_bounds__(256) void og_rms_kernel(const __bf16* ao,
    const __bf16* woutT_l, const __bf16* wgateT_l, const float* bgate_l,
    float* xw, __bf16* xn) {
  int b = blockIdx.x, tid = threadIdx.x, w = tid >> 5, lane = tid & 31, lr = lane & 15;
  __shared__ __bf16 xs[64][264];
  __shared__ float  red[256];
  __shared__ float  scale[64];
  for (int t = w; t < 64; t += 8) {
    int mt = t >> 4, nt = t & 15, col = nt * 16 + lr;
    v8f ac_o = {}, ac_g = {};
    const __bf16* arow = ao + ((size_t)b * 64 + mt * 16 + lr) * 256;
#pragma unroll
    for (int kb_ = 0; kb_ < 8; ++kb_) {
      int k0 = kb_ << 5;
      v16bf a  = afrag_bf16(arow, k0, lane);
      v16bf bo = bfrag_bf16T(woutT_l,  256, k0, col, lane);
      v16bf bg = bfrag_bf16T(wgateT_l, 256, k0, col, lane);
      ac_o = wmma_bf16(a, bo, ac_o);
      ac_g = wmma_bf16(a, bg, ac_g);
    }
    int m0 = mt * 16 + ((lane >> 4) << 3);
    float bgc = bgate_l[col];
#pragma unroll
    for (int v = 0; v < 8; ++v) {
      float g = ac_g[v] + bgc;
      float sig = 1.f / (1.f + __expf(-g));
      size_t xi = ((size_t)b * 64 + m0 + v) * 256 + col;
      float xnew = xw[xi] + ac_o[v] * sig;
      xw[xi] = xnew;
      xs[m0 + v][col] = (__bf16)xnew;
    }
  }
  __syncthreads();
  rms_epilogue(xs, red, scale, xn + (size_t)b * 64 * 256, tid);
}

// ---- fused MLP + residual + fused rmsnorm ----------------------------------
__global__ __launch_bounds__(256) void mlp_rms_kernel(const __bf16* xn_in,
    const __bf16* wmgT_l, const float* bmg_l, const __bf16* wmoT_l,
    const float* bmo_l, float* xw, __bf16* xn_out) {
  int b = blockIdx.x, tid = threadIdx.x, w = tid >> 5, lane = tid & 31, lr = lane & 15;
  __shared__ __bf16 hb[64][136];
  __shared__ __bf16 xs[64][264];
  __shared__ float  red[256];
  __shared__ float  scale[64];
  v8f accd[8];
  v8f vzero = {};
#pragma unroll
  for (int i = 0; i < 8; ++i) accd[i] = vzero;
  const __bf16* xnb = xn_in + (size_t)b * 64 * 256;
  for (int jc = 0; jc < 8; ++jc) {            // 1024 hidden in chunks of 128
#pragma unroll
    for (int i2 = 0; i2 < 4; ++i2) {
      int t2 = w + 8 * i2;
      int mt = t2 >> 3, nt = t2 & 7, cloc = nt * 16 + lr;
      int colg = jc * 128 + cloc;
      v8f ag = {}, av = {};
      const __bf16* arow = xnb + (size_t)(mt * 16 + lr) * 256;
#pragma unroll
      for (int kb_ = 0; kb_ < 8; ++kb_) {
        int k0 = kb_ << 5;
        v16bf a  = afrag_bf16(arow, k0, lane);
        v16bf bg = bfrag_bf16T(wmgT_l, 256, k0, colg, lane);
        v16bf bv = bfrag_bf16T(wmgT_l, 256, k0, 1024 + colg, lane);
        ag = wmma_bf16(a, bg, ag);
        av = wmma_bf16(a, bv, av);
      }
      int m0 = mt * 16 + ((lane >> 4) << 3);
      float bgc = bmg_l[colg], bvc = bmg_l[1024 + colg];
#pragma unroll
      for (int v = 0; v < 8; ++v) {
        float g  = ag[v] + bgc;
        float vv = av[v] + bvc;
        float gl = 0.5f * g * (1.f + erff(g * 0.7071067811865476f));
        hb[m0 + v][cloc] = (__bf16)(vv * gl);
      }
    }
    __syncthreads();
#pragma unroll
    for (int i = 0; i < 8; ++i) {
      int t = w + 8 * i;
      int mt = t >> 4, nt = t & 15, col = nt * 16 + lr;
      v8f acc = accd[i];
#pragma unroll
      for (int k2 = 0; k2 < 4; ++k2) {
        int k0 = k2 << 5;
        v16bf a  = afrag_bf16(&hb[mt * 16 + lr][0], k0, lane);
        v16bf bm = bfrag_bf16T(wmoT_l, 1024, jc * 128 + k0, col, lane);
        acc = wmma_bf16(a, bm, acc);
      }
      accd[i] = acc;
    }
    __syncthreads();
  }
#pragma unroll
  for (int i = 0; i < 8; ++i) {
    int t = w + 8 * i;
    int mt = t >> 4, nt = t & 15, col = nt * 16 + lr;
    int m0 = mt * 16 + ((lane >> 4) << 3);
    float bo = bmo_l[col];
#pragma unroll
    for (int v = 0; v < 8; ++v) {
      size_t xi = ((size_t)b * 64 + m0 + v) * 256 + col;
      float xnew = xw[xi] + accd[i][v] + bo;
      xw[xi] = xnew;
      xs[m0 + v][col] = (__bf16)xnew;
    }
  }
  __syncthreads();
  rms_epilogue(xs, red, scale, xn_out + (size_t)b * 64 * 256, tid);
}

// ---- final head + patch_out + unpatchify -----------------------------------
__global__ __launch_bounds__(256) void head_kernel(const __bf16* xn, const __bf16* wheadT,
    const float* bhead, const float* wpo, const float* bpo, float* out) {
  int b = blockIdx.x, tid = threadIdx.x, w = tid >> 5, lane = tid & 31, lr = lane & 15;
  __shared__ __bf16 hb2[64][264];
  const __bf16* xnb = xn + (size_t)b * 64 * 256;
  for (int t = w; t < 64; t += 8) {
    int mt = t >> 4, nt = t & 15, col = nt * 16 + lr;
    v8f acc = {};
#pragma unroll
    for (int kb_ = 0; kb_ < 8; ++kb_) {
      int k0 = kb_ << 5;
      v16bf a  = afrag_bf16(xnb + (size_t)(mt * 16 + lr) * 256, k0, lane);
      v16bf bm = bfrag_bf16T(wheadT, 256, k0, col, lane);  // norm_final_w pre-folded
      acc = wmma_bf16(a, bm, acc);
    }
    int m0 = mt * 16 + ((lane >> 4) << 3);
    float bh = bhead[col];
#pragma unroll
    for (int v = 0; v < 8; ++v) hb2[m0 + v][col] = (__bf16)(acc[v] + bh);
  }
  __syncthreads();
  for (int idx = tid; idx < 768; idx += 256) {
    int s = idx / 12, j = idx % 12;
    float sum = bpo[j];
    for (int c = 0; c < 256; ++c) sum += (float)hb2[s][c] * wpo[c * 12 + j];
    int gy = s >> 3, gx = s & 7, ch = j >> 2, py = (j >> 1) & 1, px = j & 1;
    out[((size_t)(b * 3 + ch) * 16 + gy * 2 + py) * 16 + gx * 2 + px] = sum;
  }
}

extern "C" void kernel_launch(void* const* d_in, const int* in_sizes, int n_in,
                              void* d_out, int out_size, void* d_ws, size_t ws_size,
                              hipStream_t stream) {
  (void)n_in; (void)out_size; (void)ws_size;
  const float* z_t    = (const float*)d_in[0];
  const float* logsnr = (const float*)d_in[1];
  const float* wpi    = (const float*)d_in[2];
  const float* bpi    = (const float*)d_in[3];
  const float* vs     = (const float*)d_in[4];
  const float* wqkv   = (const float*)d_in[5];
  const float* wout   = (const float*)d_in[6];
  const float* wgate  = (const float*)d_in[7];
  const float* bgate  = (const float*)d_in[8];
  const float* wmg    = (const float*)d_in[9];
  const float* bmg    = (const float*)d_in[10];
  const float* wmo    = (const float*)d_in[11];
  const float* bmo    = (const float*)d_in[12];
  const float* nfw    = (const float*)d_in[13];
  const float* whead  = (const float*)d_in[14];
  const float* bhead  = (const float*)d_in[15];
  const float* wpo    = (const float*)d_in[16];
  const float* bpo    = (const float*)d_in[17];
  int B = in_sizes[1];  // logsnr is [B]

  // workspace layout (256B aligned blocks)
  char* ws = (char*)d_ws;
  size_t off = 0;
  auto carve = [&](size_t bytes) { char* p = ws + off; off += (bytes + 255) & ~(size_t)255; return p; };
  __bf16* Qbf    = (__bf16*)carve((size_t)8 * 4096 * 2);
  __bf16* QTbf   = (__bf16*)carve((size_t)8 * 4096 * 2);
  __bf16* wqkvT  = (__bf16*)carve((size_t)8 * 768 * 256 * 2);
  __bf16* woutT  = (__bf16*)carve((size_t)8 * 256 * 256 * 2);
  __bf16* wgateT = (__bf16*)carve((size_t)8 * 256 * 256 * 2);
  __bf16* wmgT   = (__bf16*)carve((size_t)8 * 2048 * 256 * 2);
  __bf16* wmoT   = (__bf16*)carve((size_t)8 * 256 * 1024 * 2);
  __bf16* wheadT = (__bf16*)carve((size_t)256 * 256 * 2);
  float*  xw     = (float*)carve((size_t)B * 64 * 256 * 4);
  __bf16* xn     = (__bf16*)carve((size_t)B * 64 * 256 * 2);
  __bf16* ao     = (__bf16*)carve((size_t)B * 64 * 256 * 2);

  // one-time weight prep (bf16, transposed to K-contiguous)
  hh_kernel<<<8, 64, 0, stream>>>(vs, Qbf, QTbf);
  for (int l = 0; l < 8; ++l) {
    wcvt_kernel<<<768, 256, 0, stream>>>(wqkv + (size_t)l * 196608,
        wqkvT + (size_t)l * 196608, 256, 768, nullptr);
    wcvt_kernel<<<256, 256, 0, stream>>>(wout + (size_t)l * 65536,
        woutT + (size_t)l * 65536, 256, 256, nullptr);
    wcvt_kernel<<<256, 256, 0, stream>>>(wgate + (size_t)l * 65536,
        wgateT + (size_t)l * 65536, 256, 256, nullptr);
    wcvt_kernel<<<2048, 256, 0, stream>>>(wmg + (size_t)l * 524288,
        wmgT + (size_t)l * 524288, 256, 2048, nullptr);
    wcvt_kernel<<<1024, 256, 0, stream>>>(wmo + (size_t)l * 262144,
        wmoT + (size_t)l * 262144, 1024, 256, nullptr);
  }
  wcvt_kernel<<<256, 256, 0, stream>>>(whead, wheadT, 256, 256, nfw);

  embed_kernel<<<B, 256, 0, stream>>>(z_t, logsnr, wpi, bpi, xw, xn);
  for (int l = 0; l < 8; ++l) {
    attn_kernel<<<B * 4, 256, 0, stream>>>(xn, wqkvT + (size_t)l * 196608,
        Qbf + (size_t)l * 4096, QTbf + (size_t)l * 4096, ao,
        ((l + 1) % 4 == 0) ? 1 : 0);
    og_rms_kernel<<<B, 256, 0, stream>>>(ao, woutT + (size_t)l * 65536,
        wgateT + (size_t)l * 65536, bgate + (size_t)l * 256, xw, xn);
    mlp_rms_kernel<<<B, 256, 0, stream>>>(xn, wmgT + (size_t)l * 524288,
        bmg + (size_t)l * 2048, wmoT + (size_t)l * 262144, bmo + (size_t)l * 256,
        xw, xn);
  }
  head_kernel<<<B, 256, 0, stream>>>(xn, wheadT, bhead, wpo, bpo, (float*)d_out);
}
